// LabelCL_43602507989897
// MI455X (gfx1250) — matrix operand order repaired and checked by reading, not compiled, not run on hardware
//
#include <hip/hip_runtime.h>
#include <hip/hip_bf16.h>
#include <cmath>

typedef __attribute__((ext_vector_type(16))) _Float16 v16h;
typedef __attribute__((ext_vector_type(8)))  _Float16 v8h;
typedef __attribute__((ext_vector_type(8)))  float    v8f;

#define BB 32
#define NN 20
#define KK 50
#define DD 128
#define MM 1000   // N*K rows per batch
#define MP 1024   // padded rows per batch

#define PANEL_ROWS 64          // 4 column tiles per LDS panel
#define NPANEL     (MP / PANEL_ROWS)        // 16
#define PITCH_H    136         // 128 halves + 8 pad -> 272 B = 68 dwords (bank-conflict-free)

// ---------------- Kernel 1: L2-normalize prototype rows, write f16 (padded to 1024 rows) ----
__global__ void norm_rows_kernel(const float* __restrict__ proto, _Float16* __restrict__ ph) {
    int blk = blockIdx.x;
    int b = blk / MP, r = blk % MP;
    int t = threadIdx.x;
    __shared__ float red[DD];
    size_t outIdx = ((size_t)b * MP + r) * DD + t;
    if (r >= MM) { ph[outIdx] = (_Float16)0.f; return; }      // block-uniform branch
    float x = proto[((size_t)b * MM + r) * DD + t];
    red[t] = x * x;
    __syncthreads();
    for (int s = DD / 2; s > 0; s >>= 1) {
        if (t < s) red[t] += red[t + s];
        __syncthreads();
    }
    float inv = 1.0f / sqrtf(red[0]);
    ph[outIdx] = (_Float16)(x * inv);
}

// ---------------- Kernel 2a: class embeddings = normalize(mean over K) --------------------
__global__ void class_embed_kernel(const float* __restrict__ cls, float* __restrict__ c) {
    int blk = blockIdx.x;            // b*NN + n
    int t = threadIdx.x;
    __shared__ float red[DD];
    const float* base = cls + ((size_t)blk * KK) * DD + t;
    float s = 0.f;
    #pragma unroll 5
    for (int k = 0; k < KK; ++k) s += base[(size_t)k * DD];
    s *= (1.0f / KK);
    red[t] = s * s;
    __syncthreads();
    for (int q = DD / 2; q > 0; q >>= 1) {
        if (t < q) red[t] += red[t + q];
        __syncthreads();
    }
    float inv = 1.0f / sqrtf(red[0]);
    c[(size_t)blk * DD + t] = s * inv;
}

// ---------------- Kernel 2b: W[b] = c c^T (20x20 per batch) -------------------------------
__global__ void cosw_kernel(const float* __restrict__ c, float* __restrict__ W) {
    int b = blockIdx.x;
    const float* cb = c + (size_t)b * NN * DD;
    float* Wb = W + (size_t)b * NN * NN;
    for (int p = threadIdx.x; p < NN * NN; p += blockDim.x) {
        int n = p / NN, m = p % NN;
        const float* cn = cb + n * DD;
        const float* cm = cb + m * DD;
        float acc = 0.f;
        #pragma unroll 8
        for (int d = 0; d < DD; ++d) acc += cn[d] * cm[d];
        Wb[p] = acc;
    }
}

// Issue one 64-row column panel (16 KB) into LDS via async global->LDS copies.
// 256 threads x 4 chunks x 16 B; per-wave ASYNCcnt += 4.
__device__ __forceinline__ void issue_panel(const _Float16* phb, int panel,
                                            const _Float16* ldsbuf, int tid) {
    #pragma unroll
    for (int q = 0; q < 4; ++q) {
        int id    = q * 256 + tid;        // 0..1023
        int row   = id >> 4;              // 0..63
        int chunk = id & 15;              // 16 B chunks within a 256 B row
        unsigned long long g =
            (unsigned long long)(phb + ((size_t)(panel * PANEL_ROWS + row)) * DD + chunk * 8);
        unsigned int l =
            (unsigned int)(unsigned long long)(ldsbuf + row * PITCH_H) + chunk * 16u;
        asm volatile("global_load_async_to_lds_b128 %0, %1, off"
                     :: "v"(l), "v"(g) : "memory");
    }
}

// ---------------- Kernel 3: fused Gram (WMMA f16) + exp/weight/log reduction --------------
// 8 waves/block share async-staged LDS column panels; one 16-anchor tile per wave.
__global__ void __launch_bounds__(256)
cl_main_kernel(const _Float16* __restrict__ ph,
               const float* __restrict__ W,
               float* __restrict__ out) {
    __shared__ _Float16 lbuf[2][PANEL_ROWS * PITCH_H];   // double-buffered panels (34 KB)
    __shared__ float Wsh[NN * NN];

    const int tid  = threadIdx.x;
    const int lane = tid & 31;
    const int wave = tid >> 5;
    const int b    = blockIdx.x >> 3;
    const int tgrp = blockIdx.x & 7;
    const int tile = tgrp * 8 + wave;      // 0..63 (tile 63 is all padding; masked, no early exit)
    const int tbase = tile * 16;

    const _Float16* phb = ph + (size_t)b * MP * DD;
    const float*    Wb  = W  + (size_t)b * NN * NN;

    const int lhalf = (lane < 16) ? 0 : 1;
    const int l16   = lane & 15;
    const int rbase = lhalf ? 8 : 0;

    for (int i = tid; i < NN * NN; i += 256) Wsh[i] = Wb[i];

    union V16 { v16h v; v8h h[2]; };

    // Hoist A operand (anchor tile, all 4 K-chunks of 32).
    // 16-bit A 16x32 layout: lane m = row m; halves K=[d0,d0+8) and [d0+16,d0+24), d0=8 upper half.
    V16 a[4];
    {
        const _Float16* arow = phb + (size_t)(tbase + l16) * DD;
        #pragma unroll
        for (int kc = 0; kc < 4; ++kc) {
            int d0 = kc * 32 + (lhalf ? 8 : 0);
            a[kc].h[0] = *(const v8h*)(arow + d0);
            a[kc].h[1] = *(const v8h*)(arow + d0 + 16);
        }
    }

    // Loop-invariant per-row (per-lane) values: hoisted out of the column loop.
    float wacc[8], inacc[8], rmask[8];
    int   acls[8], rowg[8];
    #pragma unroll
    for (int r = 0; r < 8; ++r) {
        wacc[r] = 0.f; inacc[r] = 0.f;
        rowg[r] = tbase + rbase + r;
        rmask[r] = (rowg[r] < MM) ? 1.f : 0.f;
        int ac = rowg[r] / KK;
        acls[r] = (ac < NN) ? ac : (NN - 1);
    }

    issue_panel(phb, 0, lbuf[0], tid);

    for (int p = 0; p < NPANEL; ++p) {
        if (p + 1 < NPANEL) {
            issue_panel(phb, p + 1, lbuf[(p + 1) & 1], tid);
            asm volatile("s_wait_asynccnt 0x4" ::: "memory");   // panel p complete (in-order)
        } else {
            asm volatile("s_wait_asynccnt 0x0" ::: "memory");
        }
        __syncthreads();                                        // all waves' copies visible

        const _Float16* buf = lbuf[p & 1];
        #pragma unroll
        for (int q2 = 0; q2 < 4; ++q2) {
            // B = P^T from LDS: lane n = column-point row; K=[d0,d0+8) and [d0+16,d0+24), d0=8 upper half.
            const _Float16* brow = buf + (q2 * 16 + l16) * PITCH_H;
            v8f acc = {};
            #pragma unroll
            for (int kc = 0; kc < 4; ++kc) {
                int d0 = kc * 32 + (lhalf ? 8 : 0);
                V16 bm;
                bm.h[0] = *(const v8h*)(brow + d0);      // ds_load_b128
                bm.h[1] = *(const v8h*)(brow + d0 + 16);
                acc = __builtin_amdgcn_wmma_f32_16x16x32_f16(
                    false, a[kc].v, false, bm.v, (short)0, acc, false, false);
            }

            const int col_g = (p * 4 + q2) * 16 + l16;
            const float cmask = (col_g < MM) ? 1.f : 0.f;
            int cc = col_g / KK;
            cc = (cc < NN) ? cc : (NN - 1);

            #pragma unroll
            for (int r = 0; r < 8; ++r) {
                float m  = rmask[r] * cmask;
                float s  = acc[r] * 10.0f;                 // 1/T
                float e  = __expf(s);
                float w  = Wsh[acls[r] * NN + cc];
                float ds = (rowg[r] == col_g) ? 1.f : 0.f; // self term, subtracted consistently
                float cs = (acls[r] == cc)    ? 1.f : 0.f; // within-class gate
                wacc[r]  += m * (w * e - ds * e);
                inacc[r] += m * (cs * s - ds * s);
            }
        }
        __syncthreads();      // compute done before buffer (p&1) is overwritten at iter p+1
    }

    // Reduce over the 16 columns held across each 16-lane half (xor stays within the half).
    #pragma unroll
    for (int r = 0; r < 8; ++r) {
        for (int m = 1; m < 16; m <<= 1) {
            wacc[r]  += __shfl_xor(wacc[r],  m, 32);
            inacc[r] += __shfl_xor(inacc[r], m, 32);
        }
    }

    if (l16 == 0) {   // lanes 0 (rows 0..7) and 16 (rows 8..15)
        float part = 0.f;
        #pragma unroll
        for (int r = 0; r < 8; ++r) {
            if (rowg[r] < MM)
                part += logf(wacc[r]) - inacc[r] * (1.0f / (KK - 1));
        }
        const float SCALE = 0.1f / (float)(NN * KK * DD);   // ALPHA / (SELF_B*N*K*D)
        if (part != 0.f) atomicAdd(out, part * SCALE);
        else             atomicAdd(out, part * SCALE);      // keep deterministic single atomic
    }
}

extern "C" void kernel_launch(void* const* d_in, const int* in_sizes, int n_in,
                              void* d_out, int out_size, void* d_ws, size_t ws_size,
                              hipStream_t stream) {
    const float* proto = (const float*)d_in[0];   // (32, 1000, 128) f32
    const float* cls   = (const float*)d_in[1];   // (32, 20, 50, 128) f32
    float* out = (float*)d_out;                   // scalar f32

    _Float16* ph = (_Float16*)d_ws;                               // 32*1024*128 f16 = 8 MB
    size_t phBytes = (size_t)BB * MP * DD * sizeof(_Float16);
    float* c = (float*)((char*)d_ws + phBytes);                   // 32*20*128 f32
    float* W = c + (size_t)BB * NN * DD;                          // 32*20*20 f32

    hipMemsetAsync(d_out, 0, sizeof(float), stream);

    norm_rows_kernel  <<<BB * MP, DD, 0, stream>>>(proto, ph);
    class_embed_kernel<<<BB * NN, DD, 0, stream>>>(cls, c);
    cosw_kernel       <<<BB,      128, 0, stream>>>(c, W);
    cl_main_kernel    <<<BB * 8,  256, 0, stream>>>(ph, W, out);
}
